// HeteroGraphSAGE_58179626992425
// MI455X (gfx1250) — matrix operand order repaired and checked by reading, not compiled, not run on hardware
//
#include <hip/hip_runtime.h>

#define CCH 128          // channels
#define AST 264          // Abuf row stride (ushorts), padded
#define WST 264          // Wbuf row stride (ushorts), padded
#define OST 132          // Obuf row stride (floats), padded
#define EPSV 1e-5f

typedef __attribute__((ext_vector_type(16))) __bf16 v16bf;
typedef __attribute__((ext_vector_type(8)))  float  v8f;

__device__ __forceinline__ unsigned short f2bf(float f) {
    unsigned int u = __float_as_uint(f);
    unsigned int r = u + 0x7FFFu + ((u >> 16) & 1u);   // round-to-nearest-even
    return (unsigned short)(r >> 16);
}

// ---------------- zero fill ----------------
__global__ void zero_kernel(float4* __restrict__ p, int n4) {
    int i = blockIdx.x * blockDim.x + threadIdx.x;
    int stride = gridDim.x * blockDim.x;
    float4 z; z.x = 0.f; z.y = 0.f; z.z = 0.f; z.w = 0.f;
    for (; i < n4; i += stride) p[i] = z;
}

// ---------------- pack [Wl | Wr] rows into bf16, per (layer, edge_type) ----------------
// Wcat[lt][n][k], k<128 -> Wl[lt][n][k], k>=128 -> Wr[lt][n][k-128]
__global__ void pack_weights(const float* __restrict__ Wl, const float* __restrict__ Wr,
                             unsigned short* __restrict__ Wcat, int total) {
    int i = blockIdx.x * blockDim.x + threadIdx.x;
    if (i >= total) return;
    int k  = i & 255;
    int n  = (i >> 8) & 127;
    int lt = i >> 15;
    float v = (k < 128) ? Wl[((size_t)lt * 128 + n) * 128 + k]
                        : Wr[((size_t)lt * 128 + n) * 128 + (k - 128)];
    Wcat[i] = f2bf(v);
}

// ---------------- edge scatter: one wave (32 lanes) per edge ----------------
__global__ void scatter_kernel(const float* __restrict__ x, const int* __restrict__ eidx, int E,
                               float* __restrict__ msum, float* __restrict__ cnt) {
    int grp  = blockIdx.x * (blockDim.x >> 5) + (threadIdx.x >> 5);
    int lane = threadIdx.x & 31;
    if (grp >= E) return;
    int src = eidx[grp];        // row 0 of (2,E)
    int dst = eidx[E + grp];    // row 1 of (2,E)
    const float4 v = *(const float4*)(x + (size_t)src * CCH + lane * 4);
    float* base = msum + (size_t)dst * CCH + lane * 4;
    atomicAdd(base + 0, v.x);
    atomicAdd(base + 1, v.y);
    atomicAdd(base + 2, v.z);
    atomicAdd(base + 3, v.w);
    if (lane == 0) atomicAdd(cnt + dst, 1.0f);
}

// ---------------- fused: out = relu(LN( (m/cnt)@Wl.T + bias + x@Wr.T )) ----------------
// Block: 256 threads = 8 waves. BM=64 rows, BN=128 (full row -> LN fusable).
// K=256 via concat [m_scaled | x] and Wcat (128 rows x 256 cols bf16).
__launch_bounds__(256)
__global__ void gemm_ln_relu(const float* __restrict__ m, const float* __restrict__ cnt,
                             const float* __restrict__ x,
                             const unsigned short* __restrict__ Wcat,
                             const float* __restrict__ bias,
                             const float* __restrict__ gmm, const float* __restrict__ bta,
                             float* __restrict__ out, int Nrows) {
    extern __shared__ unsigned char smem[];
    unsigned short* Abuf = (unsigned short*)smem;                      // 64  x AST
    unsigned short* Wbuf = (unsigned short*)(smem + 64 * AST * 2);     // 128 x WST
    float*          Obuf = (float*)smem;                               // overlay on Abuf after K loop

    const int tid = threadIdx.x;
    const int r0  = blockIdx.x * 64;

    // Stage weights: 128x256 ushorts = 4096 uint4
    {
        const uint4* Wg = (const uint4*)Wcat;
        for (int i = tid; i < 4096; i += 256) {
            int e   = i << 3;
            int row = e >> 8;
            int col = e & 255;
            *(uint4*)(Wbuf + row * WST + col) = Wg[i];
        }
    }
    // Stage A: 64 rows x 256 cols -> bf16 (mean-scale folded into k<128 half)
    {
        for (int j = tid; j < 4096; j += 256) {      // 64 rows * 64 float4-slots
            int r    = j >> 6;
            int k0   = (j & 63) * 4;
            int node = r0 + r; if (node >= Nrows) node = Nrows - 1;
            float4 v;
            if (k0 < 128) {
                float sc = 1.0f / fmaxf(cnt[node], 1.0f);
                v = *(const float4*)(m + (size_t)node * CCH + k0);
                v.x *= sc; v.y *= sc; v.z *= sc; v.w *= sc;
            } else {
                v = *(const float4*)(x + (size_t)node * CCH + (k0 - 128));
            }
            uint2 pk;
            pk.x = (unsigned)f2bf(v.x) | ((unsigned)f2bf(v.y) << 16);
            pk.y = (unsigned)f2bf(v.z) | ((unsigned)f2bf(v.w) << 16);
            *(uint2*)(Abuf + r * AST + k0) = pk;
        }
    }
    __syncthreads();

    const int wave = tid >> 5;
    const int lane = tid & 31;
    const int half = lane >> 4;
    const int lr   = lane & 15;
    const int mt   = wave & 3;   // 4 M-tiles of 16
    const int ng   = wave >> 1 >> 1;  // wave>>2: 2 N-groups of 64

    v8f z = {0.f, 0.f, 0.f, 0.f, 0.f, 0.f, 0.f, 0.f};
    v8f acc[4] = {z, z, z, z};

    #pragma unroll
    for (int ks = 0; ks < 8; ++ks) {
        union { v16bf v; uint4 q[2]; } af;
        const unsigned short* ap = Abuf + (mt * 16 + lr) * AST + ks * 32 + half * 8;
        af.q[0] = *(const uint4*)ap;
        af.q[1] = *(const uint4*)(ap + 16);
        #pragma unroll
        for (int nt = 0; nt < 4; ++nt) {
            union { v16bf v; uint4 q[2]; } bfr;
            const unsigned short* bp = Wbuf + (ng * 64 + nt * 16 + lr) * WST + ks * 32 + half * 8;
            bfr.q[0] = *(const uint4*)bp;
            bfr.q[1] = *(const uint4*)(bp + 16);
            acc[nt] = __builtin_amdgcn_wmma_f32_16x16x32_bf16(
                false, af.v, false, bfr.v, (short)0, acc[nt], false, false);
        }
    }
    __syncthreads();   // done reading Abuf/Wbuf

    // Spill D tiles to LDS so each row is visible to one wave for LayerNorm
    #pragma unroll
    for (int nt = 0; nt < 4; ++nt) {
        int col = ng * 64 + nt * 16 + lr;
        #pragma unroll
        for (int v = 0; v < 8; ++v) {
            int row = mt * 16 + half * 8 + v;
            Obuf[row * OST + col] = acc[nt][v];
        }
    }
    __syncthreads();

    // bias + LayerNorm + ReLU, one wave per row, 4 channels per lane
    const float4 bv = *(const float4*)(bias + lane * 4);
    const float4 gv = *(const float4*)(gmm  + lane * 4);
    const float4 tv = *(const float4*)(bta  + lane * 4);
    for (int r = wave; r < 64; r += 8) {
        int node = r0 + r;
        if (node >= Nrows) continue;  // uniform across the wave
        float4 v = *(float4*)(Obuf + r * OST + lane * 4);
        v.x += bv.x; v.y += bv.y; v.z += bv.z; v.w += bv.w;
        float s  = v.x + v.y + v.z + v.w;
        float ss = v.x * v.x + v.y * v.y + v.z * v.z + v.w * v.w;
        #pragma unroll
        for (int off = 16; off >= 1; off >>= 1) {
            s  += __shfl_xor(s,  off, 32);
            ss += __shfl_xor(ss, off, 32);
        }
        float mean = s * (1.0f / 128.0f);
        float var  = ss * (1.0f / 128.0f) - mean * mean;
        float rstd = rsqrtf(var + EPSV);
        float4 o;
        o.x = fmaxf((v.x - mean) * rstd * gv.x + tv.x, 0.0f);
        o.y = fmaxf((v.y - mean) * rstd * gv.y + tv.y, 0.0f);
        o.z = fmaxf((v.z - mean) * rstd * gv.z + tv.z, 0.0f);
        o.w = fmaxf((v.w - mean) * rstd * gv.w + tv.w, 0.0f);
        *(float4*)(out + (size_t)node * CCH + lane * 4) = o;
    }
}

static inline void zero_launch(float* p, size_t nfloats, hipStream_t s) {
    int n4 = (int)(nfloats / 4);
    int blocks = (n4 + 255) / 256; if (blocks > 4096) blocks = 4096;
    zero_kernel<<<blocks, 256, 0, s>>>((float4*)p, n4);
}

extern "C" void kernel_launch(void* const* d_in, const int* in_sizes, int n_in,
                              void* d_out, int out_size, void* d_ws, size_t ws_size,
                              hipStream_t stream) {
    const float* x_user = (const float*)d_in[0];
    const float* x_item = (const float*)d_in[1];
    const int*   e_u2i  = (const int*)d_in[2];
    const int*   e_i2u  = (const int*)d_in[3];
    const float* Wl     = (const float*)d_in[4];
    const float* bl     = (const float*)d_in[5];
    const float* Wr     = (const float*)d_in[6];
    const float* gmm    = (const float*)d_in[7];
    const float* bta    = (const float*)d_in[8];
    float*       out    = (float*)d_out;

    const int NU = in_sizes[0] / CCH;
    const int NI = in_sizes[1] / CCH;
    const int E  = in_sizes[2] / 2;
    const int L  = in_sizes[4] / (2 * CCH * CCH);

    // workspace carve-out
    unsigned char* w = (unsigned char*)d_ws;
    auto carve = [&](size_t bytes) -> unsigned char* {
        unsigned char* p = w;
        w += (bytes + 255) & ~(size_t)255;
        return p;
    };
    float* bufU[2] = { (float*)carve((size_t)NU * CCH * 4), (float*)carve((size_t)NU * CCH * 4) };
    float* bufI[2] = { (float*)carve((size_t)NI * CCH * 4), (float*)carve((size_t)NI * CCH * 4) };
    float* cntU = (float*)carve((size_t)NU * 4);
    float* cntI = (float*)carve((size_t)NI * 4);
    unsigned short* Wcat = (unsigned short*)carve((size_t)L * 2 * CCH * 256 * 2);

    const int SMEM_BYTES = 64 * AST * 2 + 128 * WST * 2;  // A(bf16) + W(bf16); O overlays A

    int totW = L * 2 * CCH * 256;
    pack_weights<<<(totW + 255) / 256, 256, 0, stream>>>(Wl, Wr, Wcat, totW);

    const float* xu = x_user;
    const float* xi = x_item;
    for (int l = 0; l < L; ++l) {
        float* mu = bufU[l & 1];
        float* mi = bufI[l & 1];
        zero_launch(mu, (size_t)NU * CCH, stream);
        zero_launch(mi, (size_t)NI * CCH, stream);
        zero_launch(cntU, NU, stream);
        zero_launch(cntI, NI, stream);

        scatter_kernel<<<(E + 7) / 8, 256, 0, stream>>>(xu, e_u2i, E, mi, cntI);  // user -> item
        scatter_kernel<<<(E + 7) / 8, 256, 0, stream>>>(xi, e_i2u, E, mu, cntU);  // item -> user

        float* outU = (l == L - 1) ? out : mu;                     // in-place over mean is block-safe
        float* outI = (l == L - 1) ? (out + (size_t)NU * CCH) : mi;

        // users: aggregated via edge type 1 (i2u); LN params node type 0
        gemm_ln_relu<<<(NU + 63) / 64, 256, SMEM_BYTES, stream>>>(
            mu, cntU, xu, Wcat + (size_t)(l * 2 + 1) * CCH * 256,
            bl + (size_t)(l * 2 + 1) * CCH, gmm + (size_t)(l * 2 + 0) * CCH,
            bta + (size_t)(l * 2 + 0) * CCH, outU, NU);
        // items: aggregated via edge type 0 (u2i); LN params node type 1
        gemm_ln_relu<<<(NI + 63) / 64, 256, SMEM_BYTES, stream>>>(
            mi, cntI, xi, Wcat + (size_t)(l * 2 + 0) * CCH * 256,
            bl + (size_t)(l * 2 + 0) * CCH, gmm + (size_t)(l * 2 + 1) * CCH,
            bta + (size_t)(l * 2 + 1) * CCH, outI, NI);

        xu = outU;
        xi = outI;
    }
}
